// Attention_54013508715307
// MI455X (gfx1250) — matrix operand order repaired and checked by reading, not compiled, not run on hardware
//
#include <hip/hip_runtime.h>
#include <hip/hip_bf16.h>

typedef __attribute__((ext_vector_type(16))) _Float16 v16h;
typedef __attribute__((ext_vector_type(8)))  float    v8f;

#define B_     8
#define N_     1025
#define NVP    1032          // padded V row length (16B-aligned rows)
#define D_     512
#define H_     8
#define DH_    64
#define M_     (B_ * N_)     // 8200 rows
#define NQKV   (3 * D_)      // 1536

// padded LDS strides (halfs): byte stride 16B-aligned for b128 frag loads,
// DWORD stride maps 16 lanes to 16 distinct bank groups (conflict-free).
#define SAW 40   // 32-wide rows  -> 80B  = 20 DWORDs
#define SBW 40
#define SQW 72   // 64-wide rows  -> 144B = 36 DWORDs
#define SKW 72
#define SVW 40
#define SPW 40

union H2 { unsigned u; _Float16 h[2]; };

// ---------------------------------------------------------------- WMMA utils

__device__ __forceinline__ v8f wmma16(v16h a, v16h b, v8f c) {
  return __builtin_amdgcn_wmma_f32_16x16x32_f16(false, a, false, b, (short)0, c,
                                                false, false);
}

// A fragment: 16(M) x 32(K) f16 from row-major p[row*stride + k]
__device__ __forceinline__ v16h load_a(const _Float16* p, int stride, int lane) {
  const int row = lane & 15, hi = (lane >> 4) & 1;
  v16h a;
  const _Float16* r = p + row * stride + hi * 16;
#pragma unroll
  for (int e = 0; e < 16; ++e) a[e] = r[e];
  return a;
}

// B fragment: 32(K) x 16(N) f16 where storage is B^T row-major
__device__ __forceinline__ v16h load_b_rows(const _Float16* p, int stride, int lane) {
  const int nn = lane & 15, hi = (lane >> 4) & 1;
  v16h b;
  const _Float16* r = p + nn * stride + hi * 16;
#pragma unroll
  for (int e = 0; e < 16; ++e) b[e] = r[e];
  return b;
}

// CDNA5 async global->LDS copy, 16B per lane, tracked by ASYNCcnt
__device__ __forceinline__ void async_copy16(unsigned lds_off, const void* gptr) {
  asm volatile("global_load_async_to_lds_b128 %0, %1, off"
               :: "v"(lds_off), "v"(gptr) : "memory");
}
__device__ __forceinline__ void async_wait0() {
  asm volatile("s_wait_asynccnt 0x0" ::: "memory");
}

// ---------------------------------------------------------------- geom bias

__global__ void geom_kernel(const float* __restrict__ klein,
                            const float* __restrict__ sigma_p,
                            float* __restrict__ geom) {
  const int b = blockIdx.y;
  const int t = blockIdx.x * 256 + threadIdx.x;
  const int NP = N_ * N_;
  if (t >= NP) return;
  const int n = t / N_, m = t % N_;
  float v = 0.f;
  if (n > 0 && m > 0) {
    const float TWO_PI = 6.28318530717958647692f;
    const float PI_    = 3.14159265358979323846f;
    const float* c1 = klein + ((size_t)b * (N_ - 1) + (n - 1)) * 2;
    const float* c2 = klein + ((size_t)b * (N_ - 1) + (m - 1)) * 2;
    float a0 = c1[0], a1 = c1[1], b0 = c2[0], b1 = c2[1];
    float d0 = fabsf(a0 - b0); d0 = fminf(d0, TWO_PI - d0);
    float d1 = fabsf(a1 - b1); d1 = fminf(d1, TWO_PI - d1);
    float dtor = d0 * d0 + d1 * d1;
    float t0 = fmodf(b0 + PI_, TWO_PI);
    float t1 = fmodf(TWO_PI - b1, TWO_PI);
    float e0 = fabsf(a0 - t0); e0 = fminf(e0, TWO_PI - e0);
    float e1 = fabsf(a1 - t1); e1 = fminf(e1, TWO_PI - e1);
    float dtw = e0 * e0 + e1 * e1;
    float sg = sigma_p[0];
    v = __expf(-fminf(dtor, dtw) / (sg * sg));
  }
  geom[(size_t)b * NP + t] = v;
}

// ---------------------------------------------------------------- QKV GEMM
// q,k -> f16 [B,H,N,DH]; v -> f16 [B,H,DH,NVP] (transposed for PV B-frags)

__global__ __launch_bounds__(128) void qkv_gemm(const float* __restrict__ x,
                                                const float* __restrict__ Wqkv,
                                                _Float16* __restrict__ qb,
                                                _Float16* __restrict__ kb,
                                                _Float16* __restrict__ vb) {
  __shared__ __align__(16) _Float16 sA[64 * SAW];   // [row][k]
  __shared__ __align__(16) _Float16 sB[64 * SBW];   // [col][k]  (W^T tile)
  const int tid = threadIdx.x, lane = tid & 31, w = tid >> 5;
  const int row0 = blockIdx.x * 64;
  const int col0 = blockIdx.y * 64;
  v8f acc[4] = {};
  for (int k0 = 0; k0 < D_; k0 += 32) {
    __syncthreads();
    for (int i = tid; i < 1024; i += 128) {      // x tile, packed pair stores
      int r = i >> 4, c2 = (i & 15) * 2;
      int gr = row0 + r;
      H2 t; t.u = 0;
      if (gr < M_) {
        const float* xp = x + (size_t)gr * D_ + k0 + c2;
        t.h[0] = (_Float16)xp[0]; t.h[1] = (_Float16)xp[1];
      }
      *reinterpret_cast<unsigned*>(sA + r * SAW + c2) = t.u;
    }
    for (int i = tid; i < 1024; i += 128) {      // W tile transposed
      int c = i & 63, r2 = (i >> 6) * 2;
      const float* wp = Wqkv + (size_t)(k0 + r2) * NQKV + col0 + c;
      H2 t; t.h[0] = (_Float16)wp[0]; t.h[1] = (_Float16)wp[NQKV];
      *reinterpret_cast<unsigned*>(sB + c * SBW + r2) = t.u;
    }
    __syncthreads();
    v16h a = load_a(sA + (w * 16) * SAW, SAW, lane);
#pragma unroll
    for (int ct = 0; ct < 4; ++ct) {
      v16h bf = load_b_rows(sB + (ct * 16) * SBW, SBW, lane);
      acc[ct] = wmma16(a, bf, acc[ct]);
    }
  }

  // epilogue: which / head are uniform per block; one division per thread
  const int which = col0 >> 9;               // 0=q 1=k 2=v (block-uniform)
  const int hh    = (col0 & 511) >> 6;       // head (block-uniform)
  const int hi = (lane >> 4) & 1, nn = lane & 15;
  const int gm0 = row0 + w * 16 + hi * 8;
  int bidx0 = gm0 / N_;
  int n0    = gm0 - bidx0 * N_;
#pragma unroll
  for (int r = 0; r < 8; ++r) {
    if (gm0 + r >= M_) continue;
    int bb = bidx0, n = n0 + r;
    if (n >= N_) { bb++; n -= N_; }
    if (which == 2) {
      size_t base = ((size_t)bb * H_ + hh) * DH_ * NVP + n;
#pragma unroll
      for (int ct = 0; ct < 4; ++ct)
        vb[base + (size_t)(ct * 16 + nn) * NVP] = (_Float16)acc[ct][r];
    } else {
      _Float16* dst = which ? kb : qb;
      size_t base = (((size_t)bb * H_ + hh) * N_ + n) * DH_ + nn;
#pragma unroll
      for (int ct = 0; ct < 4; ++ct)
        dst[base + ct * 16] = (_Float16)acc[ct][r];
    }
  }
}

// ---------------------------------------------------------------- attention

__global__ __launch_bounds__(128) void attn_kernel(const _Float16* __restrict__ qb,
                                                   const _Float16* __restrict__ kb,
                                                   const _Float16* __restrict__ vb,
                                                   const float* __restrict__ geom,
                                                   const float* __restrict__ Wg,
                                                   const float* __restrict__ bg,
                                                   const float* __restrict__ alpha_p,
                                                   _Float16* __restrict__ ob) {
  __shared__ __align__(16) _Float16 sQ[4][16 * SQW];
  __shared__ __align__(16) _Float16 sK[4][32 * SKW];
  __shared__ __align__(16) _Float16 sVt[4][64 * SVW];  // [dh][key]
  __shared__ __align__(16) _Float16 sP[4][16 * SPW];
  __shared__ float sG[4][16];

  const int tid = threadIdx.x, lane = tid & 31, w = tid >> 5;
  const int h = blockIdx.y, b = blockIdx.z;
  int qt = blockIdx.x * 4 + w; if (qt > 64) qt = 64;   // keep waves uniform
  const int q0 = qt * 16;
  const size_t bh  = ((size_t)b * H_ + h) * N_;
  const size_t bhv = ((size_t)b * H_ + h) * DH_;
  const float scale = 0.125f;                          // DH^-0.5
  const int hi = (lane >> 4) & 1, nn = lane & 15;
  _Float16* sQw = sQ[w]; _Float16* sKw = sK[w];
  _Float16* sVw = sVt[w]; _Float16* sPw = sP[w];

  { // stage Q tile
    const uint4* src = (const uint4*)(qb + (bh + q0) * DH_);
    for (int i = lane; i < 128; i += 32) {
      int row = i >> 3, col = i & 7;
      *(uint4*)(sQw + row * SQW + col * 8) = src[i];
    }
  }
  __syncthreads();
  if (lane < 16) { // gate = alpha * sigmoid(q . Wg + bg)
    float g = bg[0];
    const _Float16* qr = sQw + lane * SQW;
#pragma unroll
    for (int d = 0; d < DH_; ++d) g += (float)qr[d] * Wg[d];
    sG[w][lane] = alpha_p[0] / (1.f + __expf(-g));
  }
  __syncthreads();

  v16h aQ0 = load_a(sQw, SQW, lane);       // dh 0..31
  v16h aQ1 = load_a(sQw + 32, SQW, lane);  // dh 32..63

  // loop-invariant per-row state (8 rows owned by this lane-half)
  float gt8[8]; const float* grow[8]; bool rok[8];
#pragma unroll
  for (int r = 0; r < 8; ++r) {
    int row = q0 + hi * 8 + r;
    rok[r] = row < N_;
    gt8[r] = sG[w][hi * 8 + r];
    grow[r] = geom + ((size_t)b * N_ + (rok[r] ? row : 0)) * N_;
  }

  v8f oacc[4] = {};
  float mrow[8], lrow[8];
#pragma unroll
  for (int r = 0; r < 8; ++r) { mrow[r] = -3.0e38f; lrow[r] = 0.f; }

  for (int kc = 0; kc < 33; ++kc) {
    const int k0g = kc * 32;
    { // async-stage K (32x64) and V^T (64x32) into padded LDS rows
      const char* srck = (const char*)(kb + (bh + k0g) * DH_);
      for (int i = lane; i < 256; i += 32) {
        int row = i >> 3, col = i & 7;
        async_copy16((unsigned)(size_t)(void*)(sKw + row * SKW + col * 8),
                     srck + (size_t)i * 16);
      }
      for (int i = lane; i < 256; i += 32) {
        int row = i >> 2, col = i & 3;   // row = dh, 4x16B per 32 keys
        async_copy16((unsigned)(size_t)(void*)(sVw + row * SVW + col * 8),
                     (const char*)(vb + (bhv + row) * NVP + k0g) + col * 16);
      }
      if (kc < 32) __builtin_prefetch(grow[lane & 7] + k0g + 32, 0, 1);
      async_wait0();
    }
    __syncthreads();

    // S = Q @ K^T : two 16-key tiles, K-dim 64 -> 2 WMMAs each
    v8f s0a = {}, s1a = {};
    s0a = wmma16(aQ0, load_b_rows(sKw, SKW, lane), s0a);
    s0a = wmma16(aQ1, load_b_rows(sKw + 32, SKW, lane), s0a);
    s1a = wmma16(aQ0, load_b_rows(sKw + 16 * SKW, SKW, lane), s1a);
    s1a = wmma16(aQ1, load_b_rows(sKw + 16 * SKW + 32, SKW, lane), s1a);

    const int key0 = k0g + nn;
    const int key1 = key0 + 16;
    const bool k0ok = key0 < N_, k1ok = key1 < N_;
    float s0[8], s1[8];
#pragma unroll
    for (int r = 0; r < 8; ++r) {
      float bias0 = (rok[r] && k0ok) ? grow[r][key0] : 0.f;
      float bias1 = (rok[r] && k1ok) ? grow[r][key1] : 0.f;
      s0[r] = k0ok ? (s0a[r] * scale + gt8[r] * bias0) : -3.0e38f;
      s1[r] = k1ok ? (s1a[r] * scale + gt8[r] * bias1) : -3.0e38f;
    }

#pragma unroll
    for (int r = 0; r < 8; ++r) {
      float mn = fmaxf(s0[r], s1[r]);
      mn = fmaxf(mn, __shfl_xor(mn, 1, 32));
      mn = fmaxf(mn, __shfl_xor(mn, 2, 32));
      mn = fmaxf(mn, __shfl_xor(mn, 4, 32));
      mn = fmaxf(mn, __shfl_xor(mn, 8, 32));
      float mnew = fmaxf(mrow[r], mn);
      float f = __expf(mrow[r] - mnew);
      mrow[r] = mnew;
      float p0 = __expf(s0[r] - mnew);
      float p1 = __expf(s1[r] - mnew);
      float ps = p0 + p1;
      ps += __shfl_xor(ps, 1, 32);
      ps += __shfl_xor(ps, 2, 32);
      ps += __shfl_xor(ps, 4, 32);
      ps += __shfl_xor(ps, 8, 32);
      lrow[r] = lrow[r] * f + ps;
#pragma unroll
      for (int ct = 0; ct < 4; ++ct) oacc[ct][r] *= f;
      sPw[(hi * 8 + r) * SPW + nn]      = (_Float16)p0;
      sPw[(hi * 8 + r) * SPW + 16 + nn] = (_Float16)p1;
    }
    __syncthreads();

    // O += P(16x32) @ V(32x64)
    v16h aP = load_a(sPw, SPW, lane);
#pragma unroll
    for (int ct = 0; ct < 4; ++ct) {
      v16h bV = load_b_rows(sVw + (ct * 16) * SVW, SVW, lane);
      oacc[ct] = wmma16(aP, bV, oacc[ct]);
    }
    __syncthreads();
  }

#pragma unroll
  for (int r = 0; r < 8; ++r) {
    if (!rok[r]) continue;
    const int row = q0 + hi * 8 + r;
    const float inv = 1.0f / lrow[r];
    _Float16* orow = ob + ((size_t)b * N_ + row) * (H_ * DH_) + h * DH_ + nn;
#pragma unroll
    for (int ct = 0; ct < 4; ++ct)
      orow[ct * 16] = (_Float16)(oacc[ct][r] * inv);
  }
}

// ---------------------------------------------------------------- out proj

__global__ __launch_bounds__(128) void oproj_gemm(const _Float16* __restrict__ ao,
                                                  const float* __restrict__ Wo,
                                                  const float* __restrict__ bo,
                                                  float* __restrict__ out) {
  __shared__ __align__(16) _Float16 sA[64 * SAW];
  __shared__ __align__(16) _Float16 sB[64 * SBW];   // [col][k]
  const int tid = threadIdx.x, lane = tid & 31, w = tid >> 5;
  const int row0 = blockIdx.x * 64;
  const int col0 = blockIdx.y * 64;
  v8f acc[4] = {};
  for (int k0 = 0; k0 < D_; k0 += 32) {
    __syncthreads();
    for (int i = tid; i < 1024; i += 128) {       // A already f16: b32 copies
      int r = i >> 4, c2 = (i & 15) * 2;
      int gr = row0 + r;
      unsigned u = 0;
      if (gr < M_) u = *reinterpret_cast<const unsigned*>(ao + (size_t)gr * D_ + k0 + c2);
      *reinterpret_cast<unsigned*>(sA + r * SAW + c2) = u;
    }
    for (int i = tid; i < 1024; i += 128) {       // Wo tile transposed
      int c = i & 63, r2 = (i >> 6) * 2;
      const float* wp = Wo + (size_t)(k0 + r2) * D_ + col0 + c;
      H2 t; t.h[0] = (_Float16)wp[0]; t.h[1] = (_Float16)wp[D_];
      *reinterpret_cast<unsigned*>(sB + c * SBW + r2) = t.u;
    }
    __syncthreads();
    v16h a = load_a(sA + (w * 16) * SAW, SAW, lane);
#pragma unroll
    for (int ct = 0; ct < 4; ++ct) {
      v16h bf = load_b_rows(sB + (ct * 16) * SBW, SBW, lane);
      acc[ct] = wmma16(a, bf, acc[ct]);
    }
  }
  const int hi = (lane >> 4) & 1;
#pragma unroll
  for (int ct = 0; ct < 4; ++ct) {
#pragma unroll
    for (int r = 0; r < 8; ++r) {
      int gm = row0 + w * 16 + hi * 8 + r;
      if (gm >= M_) continue;
      int c = col0 + ct * 16 + (lane & 15);
      out[(size_t)gm * D_ + c] = acc[ct][r] + bo[c];
    }
  }
}

// ---------------------------------------------------------------- launcher

extern "C" void kernel_launch(void* const* d_in, const int* in_sizes, int n_in,
                              void* d_out, int out_size, void* d_ws, size_t ws_size,
                              hipStream_t stream) {
  const float* x     = (const float*)d_in[0];
  const float* klein = (const float*)d_in[1];
  const float* Wqkv  = (const float*)d_in[2];
  const float* Wg    = (const float*)d_in[3];
  const float* bg    = (const float*)d_in[4];
  const float* Wo    = (const float*)d_in[5];
  const float* bo    = (const float*)d_in[6];
  const float* alpha = (const float*)d_in[7];
  const float* sigma = (const float*)d_in[8];
  float* out = (float*)d_out;

  // workspace: q | k | v(transposed, padded) | attn_out (f16) | geom (f32)
  const size_t nQK = (size_t)B_ * H_ * N_ * DH_;    // 4,198,400 halfs
  const size_t nV  = (size_t)B_ * H_ * DH_ * NVP;   // 4,227,072 halfs
  _Float16* qb = (_Float16*)d_ws;
  _Float16* kb = qb + nQK;
  _Float16* vb = kb + nQK;
  _Float16* ao = vb + nV;
  float* geom  = (float*)(ao + nQK);

  geom_kernel<<<dim3((N_ * N_ + 255) / 256, B_), 256, 0, stream>>>(klein, sigma, geom);
  qkv_gemm<<<dim3((M_ + 63) / 64, NQKV / 64), 128, 0, stream>>>(x, Wqkv, qb, kb, vb);
  attn_kernel<<<dim3(17, H_, B_), 128, 0, stream>>>(qb, kb, vb, geom, Wg, bg, alpha, ao);
  oproj_gemm<<<dim3((M_ + 63) / 64, D_ / 64), 128, 0, stream>>>(ao, Wo, bo, out);
}
